// WildBlock_40389872451932
// MI455X (gfx1250) — matrix operand order repaired and checked by reading, compiled, not run on hardware
//
#include <hip/hip_runtime.h>

#define DEVINL __device__ __forceinline__

typedef unsigned short     u16;
typedef unsigned int       u32;
typedef unsigned long long u64;
typedef __attribute__((ext_vector_type(16))) __bf16 v16bf;
typedef __attribute__((ext_vector_type(8)))  float  v8f;
typedef __attribute__((ext_vector_type(4)))  u32    u32x4;

union FragA { v16bf v; u32x4 q[2]; };

DEVINL u16 f2bf(float f) {
  union { float f; u32 u; } x; x.f = f;
  u32 u = x.u;
  u32 r = u + 0x7FFFu + ((u >> 16) & 1u);
  return (u16)(r >> 16);
}
DEVINL float bf2f(u16 h) {
  union { u32 u; float f; } x; x.u = ((u32)h) << 16; return x.f;
}
DEVINL float gelu_f(float x) {
  return 0.5f * x * (1.0f + tanhf(0.7978845608028654f * (x + 0.044715f * x * x * x)));
}
DEVINL v8f wmma_bf(const FragA& a, const FragA& b, v8f c) {
  return __builtin_amdgcn_wmma_f32_16x16x32_bf16(false, a.v, false, b.v, (short)0, c, false, false);
}
// Generic LDS pointer low 32 bits == LDS offset (flat aperture truncation).
DEVINL u32 ldsOff(const void* p) { return (u32)(u64)p; }
// CDNA5 async global->LDS copy, 16B per lane, tracked by ASYNCcnt (in-order).
DEVINL void async_b128(u32 loff, const void* gsrc) {
  asm volatile("global_load_async_to_lds_b128 %0, %1, off"
               :: "v"(loff), "v"(gsrc) : "memory");
}
template<int Np>
DEVINL void wait_async() {
  asm volatile("s_wait_asynccnt %0" :: "i"(Np) : "memory");
}

// ---------------------------------------------------------------- weight convert+transpose
// dst[e][n][k] = bf16(src[e][k][n]); rows=K, cols=N per matrix.
__global__ __launch_bounds__(256)
void cvt_transpose_k(const float* __restrict__ src, u16* __restrict__ dst,
                     int rows, int cols, long total) {
  long i = (long)blockIdx.x * 256 + threadIdx.x;
  if (i >= total) return;
  long per = (long)rows * cols;
  long e = i / per, rem = i % per;
  int n = (int)(rem / rows), k = (int)(rem % rows);
  dst[i] = f2bf(src[e * per + (long)k * cols + n]);
}

// ---------------------------------------------------------------- rmsnorm -> bf16
__global__ __launch_bounds__(256)
void rmsnorm_k(const float* __restrict__ x, const float* __restrict__ w,
               u16* __restrict__ out, int D) {
  int tok = blockIdx.x, tid = threadIdx.x;
  const float* row = x + (long)tok * D;
  __shared__ float red[256];
  float s = 0.f;
  for (int j = tid; j < D; j += 256) { float v = row[j]; s += v * v; }
  red[tid] = s; __syncthreads();
  for (int off = 128; off > 0; off >>= 1) {
    if (tid < off) red[tid] += red[tid + off];
    __syncthreads();
  }
  float inv = rsqrtf(red[0] / (float)D + 1e-6f);
  u16* orow = out + (long)tok * D;
  for (int j = tid; j < D; j += 256) orow[j] = f2bf(row[j] * inv * w[j]);
}

// ---------------------------------------------------------------- WMMA GEMM
// Block 256 thr = 8 waves (4x2). Block tile 128x128, wave tile 32x64 (8 WMMA / 32-k).
// A bf16 [row][k]; B bf16 pre-transposed [n][k]. Both staged via
// global_load_async_to_lds_b128; B double-buffered so the next weight tile
// streams in (ASYNCcnt<=4 in flight) while WMMAs run on the current one.
template<bool MOE, bool GATHER_A, bool C_BF16, bool GELU, bool RESID>
__global__ __launch_bounds__(256)
void gemm_k(const u16* __restrict__ A_, long lda,
            const u16* __restrict__ Bt, long ldbt, long strideB,
            void* __restrict__ C_, long ldc,
            int M, int Kd,
            const float* __restrict__ resid,
            const int* __restrict__ seg, const int* __restrict__ perm) {
  int tid = threadIdx.x, lane = tid & 31, wv = tid >> 5;
  int wm = wv >> 1, wn = wv & 1;
  int half = lane >> 4, l16 = lane & 15;

  int blockN = blockIdx.x * 128;
  int blockM, Mvalid, segbase = 0;
  const u16* Bw;
  if (MOE) {
    int e = blockIdx.z;
    segbase = seg[e];
    Mvalid = seg[e + 1] - segbase;
    blockM = blockIdx.y * 128;
    if (blockM >= Mvalid) return;
    Bw = Bt + (long)e * strideB;
  } else {
    blockM = blockIdx.y * 128; Mvalid = M; Bw = Bt;
  }

  __shared__ alignas(16) u16 As[128][72];
  __shared__ alignas(16) u16 Bs[2][128][72];

  v8f acc[2][4];
#pragma unroll
  for (int i = 0; i < 2; i++)
#pragma unroll
    for (int t = 0; t < 4; t++) acc[i][t] = {};

  // per-thread chunk row/col resolution (k-invariant)
  long arowOff[4]; bool avalid[4]; long bcolOff[4];
#pragma unroll
  for (int i = 0; i < 4; i++) {
    int idx = tid + 256 * i;
    int r = idx >> 3;
    int gr = blockM + r;
    avalid[i] = gr < Mvalid;
    long arow = 0;
    if (avalid[i]) {
      if (MOE) arow = GATHER_A ? (long)perm[segbase + gr] : (long)(segbase + gr);
      else     arow = gr;
    }
    arowOff[i] = arow * lda;
    bcolOff[i] = (long)(blockN + r) * ldbt;
  }

  // invalid A rows: zero once (asyncs never touch them)
  {
    u32x4 zero = {0, 0, 0, 0};
#pragma unroll
    for (int i = 0; i < 4; i++) {
      if (!avalid[i]) {
        int idx = tid + 256 * i;
        int r = idx >> 3, c8 = idx & 7;
        *(u32x4*)&As[r][c8 * 8] = zero;
      }
    }
  }

  // prologue: B stage 0 in flight
#pragma unroll
  for (int i = 0; i < 4; i++) {
    int idx = tid + 256 * i;
    int r = idx >> 3, c8 = idx & 7;
    async_b128(ldsOff(&Bs[0][r][c8 * 8]), Bw + bcolOff[i] + c8 * 8);
  }

  const int nStages = Kd >> 6;
  for (int st = 0; st < nStages; st++) {
    const int k0 = st * 64;
    // A stage st (single buffer: bottom barrier of st-1 protects overwrite)
#pragma unroll
    for (int i = 0; i < 4; i++) {
      if (avalid[i]) {
        int idx = tid + 256 * i;
        int r = idx >> 3, c8 = idx & 7;
        async_b128(ldsOff(&As[r][c8 * 8]), A_ + arowOff[i] + k0 + c8 * 8);
      }
    }
    bool notLast = (st + 1 < nStages);
    if (notLast) {
      // B stage st+1 into the other buffer; stays in flight during compute
#pragma unroll
      for (int i = 0; i < 4; i++) {
        int idx = tid + 256 * i;
        int r = idx >> 3, c8 = idx & 7;
        async_b128(ldsOff(&Bs[(st + 1) & 1][r][c8 * 8]),
                   Bw + bcolOff[i] + k0 + 64 + c8 * 8);
      }
      __builtin_prefetch(Bw + bcolOff[0] + k0 + 128, 0, 1);
      wait_async<4>();   // drain B(st) + A(st); leave B(st+1) in flight
    } else {
      wait_async<0>();
    }
    __syncthreads();

    const u16 (*Bsc)[72] = Bs[st & 1];
#pragma unroll
    for (int s = 0; s < 2; s++) {
      FragA a0, a1, b[4];
      const u16* ap0 = &As[wm * 32 + l16][0];
      a0.q[0] = *(const u32x4*)(ap0 + s * 32 + half * 8);
      a0.q[1] = *(const u32x4*)(ap0 + s * 32 + 16 + half * 8);
      const u16* ap1 = &As[wm * 32 + 16 + l16][0];
      a1.q[0] = *(const u32x4*)(ap1 + s * 32 + half * 8);
      a1.q[1] = *(const u32x4*)(ap1 + s * 32 + 16 + half * 8);
#pragma unroll
      for (int t = 0; t < 4; t++) {
        const u16* bp = &Bsc[wn * 64 + t * 16 + l16][0];
        b[t].q[0] = *(const u32x4*)(bp + s * 32 + half * 8);
        b[t].q[1] = *(const u32x4*)(bp + s * 32 + 16 + half * 8);
      }
#pragma unroll
      for (int t = 0; t < 4; t++) {
        acc[0][t] = wmma_bf(a0, b[t], acc[0][t]);
        acc[1][t] = wmma_bf(a1, b[t], acc[1][t]);
      }
    }
    __syncthreads();
  }

#pragma unroll
  for (int i = 0; i < 2; i++) {
    int rowLocal = wm * 32 + i * 16 + 8 * half;
#pragma unroll
    for (int rr = 0; rr < 8; rr++) {
      int r = blockM + rowLocal + rr;
      if (r < Mvalid) {
        long crow = MOE ? (long)(segbase + r) : (long)r;
#pragma unroll
        for (int t = 0; t < 4; t++) {
          int col = blockN + wn * 64 + t * 16 + l16;
          float v = acc[i][t][rr];
          if (GELU) v = gelu_f(v);
          if (RESID) v += resid[crow * ldc + col];
          if (C_BF16) ((u16*)C_)[crow * ldc + col] = f2bf(v);
          else        ((float*)C_)[crow * ldc + col] = v;
        }
      }
    }
  }
}

// ---------------------------------------------------------------- flash attention (bf16 q/k/v)
__global__ __launch_bounds__(256)
void attn_k(const u16* __restrict__ qb, const u16* __restrict__ kb,
            const u16* __restrict__ vb, u16* __restrict__ ob) {
  const int T = 1024;
  const float SCALE = 0.125f;
  const int bT = blockIdx.z * T;
  const int h = blockIdx.y;
  const int qBase = blockIdx.x * 128;
  int tid = threadIdx.x, lane = tid & 31, wv = tid >> 5;
  int half = lane >> 4, l16 = lane & 15;

  __shared__ alignas(16) u16 Qs[128][72];
  __shared__ alignas(16) u16 Ks[64][72];   // [kj][d]
  __shared__ alignas(16) u16 Vt[64][72];   // [d][kj]
  __shared__ alignas(16) u16 Pl[8][16][72];

  { // stage Q tile via async copies
#pragma unroll
    for (int i = 0; i < 4; i++) {
      int idx = tid + 256 * i;
      int r = idx >> 3, c8 = idx & 7;
      async_b128(ldsOff(&Qs[r][c8 * 8]),
                 qb + (long)(bT + qBase + r) * 768 + h * 64 + c8 * 8);
    }
    wait_async<0>();
  }
  __syncthreads();

  FragA aq[2];
  {
    const u16* rp = &Qs[wv * 16 + l16][0];
#pragma unroll
    for (int ks = 0; ks < 2; ks++) {
      aq[ks].q[0] = *(const u32x4*)(rp + ks * 32 + half * 8);
      aq[ks].q[1] = *(const u32x4*)(rp + ks * 32 + 16 + half * 8);
    }
  }

  v8f oAcc[4]; float m[8], l[8];
#pragma unroll
  for (int t = 0; t < 4; t++) oAcc[t] = {};
#pragma unroll
  for (int rr = 0; rr < 8; rr++) { m[rr] = -INFINITY; l[rr] = 0.f; }

  const int rowA = qBase + wv * 16 + 8 * half;
  const int ktMax = (qBase + 127) >> 6;

  for (int kt = 0; kt <= ktMax; kt++) {
    __syncthreads();
    { // K via async; V transposed via LDS stores
#pragma unroll
      for (int i = 0; i < 2; i++) {
        int idx = tid + 256 * i;
        int r = idx >> 3, c8 = idx & 7;
        long rowoff = (long)(bT + kt * 64 + r) * 768 + h * 64 + c8 * 8;
        async_b128(ldsOff(&Ks[r][c8 * 8]), kb + rowoff);
        union { u32x4 q; u16 s[8]; } vv;
        vv.q = *(const u32x4*)(vb + rowoff);
#pragma unroll
        for (int j = 0; j < 8; j++) Vt[c8 * 8 + j][r] = vv.s[j];
      }
      wait_async<0>();
    }
    __syncthreads();

    v8f sAcc[4];
#pragma unroll
    for (int t = 0; t < 4; t++) sAcc[t] = {};
#pragma unroll
    for (int ks = 0; ks < 2; ks++) {
#pragma unroll
      for (int t = 0; t < 4; t++) {
        FragA b;
        const u16* kp = &Ks[t * 16 + l16][0];
        b.q[0] = *(const u32x4*)(kp + ks * 32 + half * 8);
        b.q[1] = *(const u32x4*)(kp + ks * 32 + 16 + half * 8);
        sAcc[t] = wmma_bf(aq[ks], b, sAcc[t]);
      }
    }

    const int colBase = kt * 64;
#pragma unroll
    for (int rr = 0; rr < 8; rr++) {
      int row = rowA + rr;
      float xv[4], mx = -INFINITY;
#pragma unroll
      for (int t = 0; t < 4; t++) {
        int col = colBase + t * 16 + l16;
        xv[t] = (col <= row) ? sAcc[t][rr] * SCALE : -INFINITY;
        mx = fmaxf(mx, xv[t]);
      }
#pragma unroll
      for (int msk = 1; msk < 16; msk <<= 1) mx = fmaxf(mx, __shfl_xor(mx, msk, 32));
      float mnew = fmaxf(m[rr], mx);
      float alpha, rsum = 0.f;
      if (mnew == -INFINITY) {
        alpha = 1.f;
#pragma unroll
        for (int t = 0; t < 4; t++) xv[t] = 0.f;
      } else {
        alpha = __expf(m[rr] - mnew);
#pragma unroll
        for (int t = 0; t < 4; t++) { xv[t] = __expf(xv[t] - mnew); rsum += xv[t]; }
      }
#pragma unroll
      for (int msk = 1; msk < 16; msk <<= 1) rsum += __shfl_xor(rsum, msk, 32);
      m[rr] = mnew;
      l[rr] = l[rr] * alpha + rsum;
#pragma unroll
      for (int t = 0; t < 4; t++) oAcc[t][rr] *= alpha;
      int prow = 8 * half + rr;
#pragma unroll
      for (int t = 0; t < 4; t++) Pl[wv][prow][t * 16 + l16] = f2bf(xv[t]);
    }
    __syncthreads();

#pragma unroll
    for (int ks = 0; ks < 2; ks++) {
      FragA ap;
      const u16* pp = &Pl[wv][l16][0];
      ap.q[0] = *(const u32x4*)(pp + ks * 32 + half * 8);
      ap.q[1] = *(const u32x4*)(pp + ks * 32 + 16 + half * 8);
#pragma unroll
      for (int t = 0; t < 4; t++) {
        FragA b;
        const u16* vp = &Vt[t * 16 + l16][0];
        b.q[0] = *(const u32x4*)(vp + ks * 32 + half * 8);
        b.q[1] = *(const u32x4*)(vp + ks * 32 + 16 + half * 8);
        oAcc[t] = wmma_bf(ap, b, oAcc[t]);
      }
    }
  }

#pragma unroll
  for (int rr = 0; rr < 8; rr++) {
    int row = rowA + rr;
    float inv = 1.0f / l[rr];
    long base = (long)(bT + row) * 768 + h * 64;
#pragma unroll
    for (int t = 0; t < 4; t++) ob[base + t * 16 + l16] = f2bf(oAcc[t][rr] * inv);
  }
}

// ---------------------------------------------------------------- router
__global__ __launch_bounds__(256)
void router_k(const u16* __restrict__ h2, const float* __restrict__ Wg,
              int* __restrict__ eidx, float* __restrict__ gates) {
  int tok = blockIdx.x, wv = threadIdx.x >> 5, lane = threadIdx.x & 31;
  __shared__ float lg[8];
  {
    float s = 0.f;
    const u16* row = h2 + (long)tok * 768;
    for (int j = lane; j < 768; j += 32) s += bf2f(row[j]) * Wg[j * 8 + wv];
#pragma unroll
    for (int msk = 16; msk > 0; msk >>= 1) s += __shfl_xor(s, msk, 32);
    if (lane == 0) lg[wv] = s;
  }
  __syncthreads();
  if (threadIdx.x == 0) {
    int i0 = 0; float v0 = lg[0];
    for (int e = 1; e < 8; e++) if (lg[e] > v0) { v0 = lg[e]; i0 = e; }
    int i1 = (i0 == 0) ? 1 : 0; float v1 = lg[i1];
    for (int e = 0; e < 8; e++) if (e != i0 && lg[e] > v1) { v1 = lg[e]; i1 = e; }
    float e1 = __expf(v1 - v0);
    float inv = 1.0f / (1.0f + e1);
    eidx[tok * 2] = i0; eidx[tok * 2 + 1] = i1;
    gates[tok * 2] = inv; gates[tok * 2 + 1] = e1 * inv;
  }
}

// ---------------------------------------------------------------- routing gather (1 block)
__global__ __launch_bounds__(256)
void route_gather_k(const int* __restrict__ eidx, int* __restrict__ seg,
                    int* __restrict__ perm, int* __restrict__ arow, int nAssign) {
  __shared__ int cnt[8], base[9], pos[8];
  int tid = threadIdx.x;
  if (tid < 8) cnt[tid] = 0;
  __syncthreads();
  for (int t = tid; t < nAssign; t += 256) atomicAdd(&cnt[eidx[t]], 1);
  __syncthreads();
  if (tid == 0) {
    int a = 0;
    for (int e = 0; e < 8; e++) { base[e] = a; a += cnt[e]; }
    base[8] = a;
  }
  __syncthreads();
  if (tid < 8) pos[tid] = base[tid];
  if (tid < 9) seg[tid] = base[tid];
  __syncthreads();
  for (int t = tid; t < nAssign; t += 256) {
    int e = eidx[t];
    int p = atomicAdd(&pos[e], 1);
    perm[p] = t >> 1;
    arow[t] = p;
  }
}

// ---------------------------------------------------------------- final combine
__global__ __launch_bounds__(256)
void combine_k(const float* __restrict__ x1, const float* __restrict__ eout,
               const int* __restrict__ arow, const float* __restrict__ gates,
               float* __restrict__ out, long total) {
  long i = (long)blockIdx.x * 256 + threadIdx.x;
  if (i >= total) return;
  int t = (int)(i / 768), d = (int)(i % 768);
  int a0 = arow[t * 2], a1 = arow[t * 2 + 1];
  out[i] = x1[i] + gates[t * 2] * eout[(long)a0 * 768 + d]
                 + gates[t * 2 + 1] * eout[(long)a1 * 768 + d];
}

// ================================================================ launch
extern "C" void kernel_launch(void* const* d_in, const int* in_sizes, int n_in,
                              void* d_out, int out_size, void* d_ws, size_t ws_size,
                              hipStream_t stream) {
  const int Bb = 2, T = 1024, D = 768, H = 12, DL = 256, E = 8, F = 3072;
  const int N = Bb * T;

  const float* x    = (const float*)d_in[0];
  const float* ln1  = (const float*)d_in[1];
  const float* ln2  = (const float*)d_in[2];
  const float* Wq   = (const float*)d_in[3];
  const float* Wdkv = (const float*)d_in[4];
  const float* Wuk  = (const float*)d_in[5];
  const float* Wuv  = (const float*)d_in[6];
  const float* Wo   = (const float*)d_in[7];
  const float* Wg   = (const float*)d_in[8];
  const float* W1   = (const float*)d_in[9];
  const float* W2   = (const float*)d_in[10];

  char* wp = (char*)d_ws;
  auto alloc = [&](size_t bytes) -> char* {
    char* p = wp; wp += (bytes + 255) & ~(size_t)255; return p;
  };
  u16* WqT   = (u16*)alloc((size_t)768 * D * 2);
  u16* WdkvT = (u16*)alloc((size_t)DL * D * 2);
  u16* WukT  = (u16*)alloc((size_t)768 * DL * 2);
  u16* WuvT  = (u16*)alloc((size_t)768 * DL * 2);
  u16* WoT   = (u16*)alloc((size_t)D * 768 * 2);
  u16* W1T   = (u16*)alloc((size_t)E * F * D * 2);
  u16* W2T   = (u16*)alloc((size_t)E * D * F * 2);
  u16* hbf   = (u16*)alloc((size_t)N * D * 2);
  u16* qbf   = (u16*)alloc((size_t)N * 768 * 2);
  u16* ckvbf = (u16*)alloc((size_t)N * DL * 2);
  u16* kbf   = (u16*)alloc((size_t)N * 768 * 2);
  u16* vbf   = (u16*)alloc((size_t)N * 768 * 2);
  u16* attbf = (u16*)alloc((size_t)N * 768 * 2);
  u16* h2bf  = (u16*)alloc((size_t)N * D * 2);
  u16* mid   = (u16*)alloc((size_t)N * 2 * F * 2);
  float* x1   = (float*)alloc((size_t)N * D * 4);
  float* eout = (float*)alloc((size_t)N * 2 * D * 4);
  float* gates = (float*)alloc((size_t)N * 2 * 4);
  int*   eidx  = (int*)alloc((size_t)N * 2 * 4);
  int*   seg   = (int*)alloc(16 * 4);
  int*   perm  = (int*)alloc((size_t)N * 2 * 4);
  int*   arow  = (int*)alloc((size_t)N * 2 * 4);

  auto cvt = [&](const float* src, u16* dst, int rows, int cols, long total) {
    cvt_transpose_k<<<(int)((total + 255) / 256), 256, 0, stream>>>(src, dst, rows, cols, total);
  };
  cvt(Wq,   WqT,   D,  768, (long)D * 768);
  cvt(Wdkv, WdkvT, D,  DL,  (long)D * DL);
  cvt(Wuk,  WukT,  DL, 768, (long)DL * 768);
  cvt(Wuv,  WuvT,  DL, 768, (long)DL * 768);
  cvt(Wo,   WoT,   768, D,  (long)768 * D);
  cvt(W1,   W1T,   D,  F,   (long)E * D * F);
  cvt(W2,   W2T,   F,  D,   (long)E * F * D);

  rmsnorm_k<<<N, 256, 0, stream>>>(x, ln1, hbf, D);
  gemm_k<false,false,true,false,false><<<dim3(6, 16), 256, 0, stream>>>(
      hbf, D, WqT, D, 0, qbf, 768, N, D, nullptr, nullptr, nullptr);
  gemm_k<false,false,true,false,false><<<dim3(2, 16), 256, 0, stream>>>(
      hbf, D, WdkvT, D, 0, ckvbf, DL, N, D, nullptr, nullptr, nullptr);
  gemm_k<false,false,true,false,false><<<dim3(6, 16), 256, 0, stream>>>(
      ckvbf, DL, WukT, DL, 0, kbf, 768, N, DL, nullptr, nullptr, nullptr);
  gemm_k<false,false,true,false,false><<<dim3(6, 16), 256, 0, stream>>>(
      ckvbf, DL, WuvT, DL, 0, vbf, 768, N, DL, nullptr, nullptr, nullptr);
  attn_k<<<dim3(8, H, Bb), 256, 0, stream>>>(qbf, kbf, vbf, attbf);
  gemm_k<false,false,false,false,true><<<dim3(6, 16), 256, 0, stream>>>(
      attbf, 768, WoT, 768, 0, x1, D, N, 768, x, nullptr, nullptr);
  rmsnorm_k<<<N, 256, 0, stream>>>(x1, ln2, h2bf, D);
  router_k<<<N, 256, 0, stream>>>(h2bf, Wg, eidx, gates);
  route_gather_k<<<1, 256, 0, stream>>>(eidx, seg, perm, arow, N * 2);
  gemm_k<true,true,true,true,false><<<dim3(24, 16, E), 256, 0, stream>>>(
      h2bf, D, W1T, D, (long)D * F, mid, F, 0, D, nullptr, seg, perm);
  gemm_k<true,false,false,false,false><<<dim3(6, 16, E), 256, 0, stream>>>(
      mid, F, W2T, F, (long)F * D, eout, D, 0, F, nullptr, seg, perm);
  long total = (long)N * D;
  combine_k<<<(int)((total + 255) / 256), 256, 0, stream>>>(
      x1, eout, arow, gates, (float*)d_out, total);
}
